// SATMessagePassing_14577119003010
// MI455X (gfx1250) — matrix operand order repaired and checked by reading, not compiled, not run on hardware
//
#include <hip/hip_runtime.h>
#include <math.h>

// ---------------------------------------------------------------------------
// Problem constants (from reference)
// ---------------------------------------------------------------------------
static const int VN     = 100000;   // variables
static const int CN     = 400000;   // clauses
static const int INW    = 96;       // input feature width
static const int OUTW   = 64;       // per-head / hidden width
static const int HD     = 256;      // H * OUT
static const int GW     = 192;      // 3 * OUT (GRU gate width)
static const int EHALF  = 1000000;  // edges per direction
static const int ETOT   = 2000000;  // total edges (row stride of edge_index)

typedef __attribute__((ext_vector_type(16))) _Float16 v16h;
typedef __attribute__((ext_vector_type(8)))  float    v8f;

// ---------------------------------------------------------------------------
// Weight pre-pack: f32 weights -> f16, swizzled into a contiguous-k WMMA B-fragment
// layout.  Fragment (c = K-chunk, tile = 16-col tile, lane) is 16 contiguous
// halves at Wp[((c*ntile + tile)*32 + lane)*16]:
//   n = tile*16 + (lane&15), kb = (lane>>4)*16, VGPR v holds K = kb+2v, +1.
//   transB==0 : B[k,o] = W[k*ldw + o]    (W stored Kin x Kout)
//   transB==1 : B[k,o] = W[o*ldw + k]    (W stored Kout x Kin, GRU W.T)
// ---------------------------------------------------------------------------
__global__ void pack_w_kernel(const float* __restrict__ W, int ldw, int transB,
                              _Float16* __restrict__ Wp, int Kin, int Kout)
{
    int idx = blockIdx.x * blockDim.x + threadIdx.x;
    int ntile = Kout >> 4;
    int total = (Kin >> 5) * ntile * 32;
    if (idx >= total) return;
    int lane = idx & 31;
    int tile = (idx >> 5) % ntile;
    int c    = (idx >> 5) / ntile;
    int n  = tile * 16 + (lane & 15);
    int kb = (lane >> 4) * 16;
    _Float16* dst = Wp + (size_t)idx * 16;
#pragma unroll
    for (int v = 0; v < 8; ++v) {
        int k = c * 32 + kb + 2 * v;
        float w0, w1;
        if (transB) {
            w0 = W[(size_t)n * ldw + k];
            w1 = W[(size_t)n * ldw + k + 1];
        } else {
            w0 = W[(size_t)k       * ldw + n];
            w1 = W[(size_t)(k + 1) * ldw + n];
        }
        dst[2 * v]     = (_Float16)w0;
        dst[2 * v + 1] = (_Float16)w1;
    }
}

// ---------------------------------------------------------------------------
// WMMA GEMM:  Y[N x (NT*64)] = X[N x KIN] @ B + bias   (f16 in, f32 accum)
// B pre-packed by pack_w_kernel. Block = 4 waves x NT tiles = full row width.
// One A fragment per chunk feeds NT WMMAs (register reuse). grid = N/16.
// ---------------------------------------------------------------------------
template<int KIN, int NT>
__global__ __launch_bounds__(128)
void gemm_wmma_t(const float* __restrict__ X, int ldx,
                 const _Float16* __restrict__ Wp,
                 const float* __restrict__ bias,
                 float* __restrict__ Y, int ldy)
{
    const int NTILE = NT * 4;            // total 16-col tiles = Kout/16
    int lane = threadIdx.x & 31;
    int wave = threadIdx.x >> 5;
    int row0 = blockIdx.x * 16;

    int n   = lane & 15;
    int hi  = lane >> 4;
    int kbA = hi * 16;                   // A: hi half-wave holds K+16..K+31

    v8f acc[NT];
#pragma unroll
    for (int t = 0; t < NT; ++t) {
        float bv = bias ? bias[(wave * NT + t) * 16 + n] : 0.0f;
#pragma unroll
        for (int r = 0; r < 8; ++r) acc[t][r] = bv;
    }

    const float* xrow = X + (size_t)(row0 + n) * ldx;
    const v16h*  Wf   = (const v16h*)Wp;

    // Prefetch next row-panel of A (speculative; dropped if out of range).
    __builtin_prefetch(xrow + (size_t)16 * ldx, 0, 1);

#pragma unroll
    for (int c = 0; c < KIN / 32; ++c) {
        // A 16x32 fragment: two contiguous 8-float runs per lane.
        const float4* p0 = (const float4*)(xrow + c * 32 + kbA);
        const float4* p1 = (const float4*)(xrow + c * 32 + 8 + kbA);
        float4 a0 = p0[0], a1 = p0[1];
        float4 a2 = p1[0], a3 = p1[1];
        v16h a;
        a[0]  = (_Float16)a0.x; a[1]  = (_Float16)a0.y;
        a[2]  = (_Float16)a0.z; a[3]  = (_Float16)a0.w;
        a[4]  = (_Float16)a1.x; a[5]  = (_Float16)a1.y;
        a[6]  = (_Float16)a1.z; a[7]  = (_Float16)a1.w;
        a[8]  = (_Float16)a2.x; a[9]  = (_Float16)a2.y;
        a[10] = (_Float16)a2.z; a[11] = (_Float16)a2.w;
        a[12] = (_Float16)a3.x; a[13] = (_Float16)a3.y;
        a[14] = (_Float16)a3.z; a[15] = (_Float16)a3.w;
#pragma unroll
        for (int t = 0; t < NT; ++t) {
            v16h b = Wf[((size_t)c * NTILE + wave * NT + t) * 32 + lane];
            acc[t] = __builtin_amdgcn_wmma_f32_16x16x32_f16(
                         false, a, false, b, (short)0, acc[t], false, false);
        }
    }

#pragma unroll
    for (int t = 0; t < NT; ++t) {
        int col0 = (wave * NT + t) * 16;
#pragma unroll
        for (int r = 0; r < 8; ++r) {
            int m = row0 + r + hi * 8;
            Y[(size_t)m * ldy + col0 + n] = acc[t][r];
        }
    }
}

// ---------------------------------------------------------------------------
// Utility kernels
// ---------------------------------------------------------------------------
__global__ void fill_kernel(float* __restrict__ p, float v, long long n)
{
    long long i = (long long)blockIdx.x * blockDim.x + threadIdx.x;
    if (i < n) p[i] = v;
}

__device__ inline void atomicMaxF(float* addr, float val)
{
    unsigned int* ua  = (unsigned int*)addr;
    unsigned int  old = *ua;
    while (__uint_as_float(old) < val) {
        unsigned int assumed = old;
        old = atomicCAS(ua, assumed, __float_as_uint(val));
        if (old == assumed) break;
    }
}

// ---------------------------------------------------------------------------
// GATv2 edge pass 1: raw attention logits + segment max.
// One wave per edge; lane covers 8 contiguous dims (head = lane/8).
// ---------------------------------------------------------------------------
__global__ __launch_bounds__(256)
void edge_score_kernel(const int* __restrict__ src, int srcSub,
                       const int* __restrict__ dst, int dstSub,
                       const float* __restrict__ xl, const float* __restrict__ xr,
                       const float* __restrict__ att,
                       float* __restrict__ alpha, float* __restrict__ amax, int nE)
{
    int lane = threadIdx.x & 31;
    int e = blockIdx.x * 8 + (threadIdx.x >> 5);
    if (e >= nE) return;
    int j = src[e] - srcSub;
    int i = dst[e] - dstSub;
    int d0 = lane * 8;
    const float* pl = xl + (size_t)j * HD + d0;
    const float* pr = xr + (size_t)i * HD + d0;
    int h = lane >> 3;
    const float* pa = att + h * OUTW + (lane & 7) * 8;
    float s = 0.f;
#pragma unroll
    for (int t = 0; t < 8; ++t) {
        float v = pl[t] + pr[t];
        v = (v > 0.f) ? v : 0.2f * v;          // leaky_relu(0.2)
        s += v * pa[t];
    }
    s += __shfl_xor(s, 1);
    s += __shfl_xor(s, 2);
    s += __shfl_xor(s, 4);
    if ((lane & 7) == 0) {
        alpha[(size_t)e * 4 + h] = s;
        atomicMaxF(&amax[(size_t)i * 4 + h], s);
    }
}

// Pass 2: p = exp(a - amax[dst]); segment-sum denom. One thread per (edge,head).
__global__ void edge_softmax_kernel(const int* __restrict__ dst, int dstSub,
                                    float* __restrict__ alpha,
                                    const float* __restrict__ amax,
                                    float* __restrict__ denom, int nE)
{
    int t = blockIdx.x * blockDim.x + threadIdx.x;
    if (t >= nE * 4) return;
    int e = t >> 2, h = t & 3;
    int i = dst[e] - dstSub;
    float p = expf(alpha[t] - amax[(size_t)i * 4 + h]);
    alpha[t] = p;
    atomicAdd(&denom[(size_t)i * 4 + h], p);
}

// Pass 3: m[dst] += xl[src] * (p / denom). One wave per edge.
__global__ __launch_bounds__(256)
void edge_aggregate_kernel(const int* __restrict__ src, int srcSub,
                           const int* __restrict__ dst, int dstSub,
                           const float* __restrict__ xl,
                           const float* __restrict__ alpha,
                           const float* __restrict__ denom,
                           float* __restrict__ m, int nE)
{
    int lane = threadIdx.x & 31;
    int e = blockIdx.x * 8 + (threadIdx.x >> 5);
    if (e >= nE) return;
    int j = src[e] - srcSub;
    int i = dst[e] - dstSub;
    int h = lane >> 3;
    float sc = alpha[(size_t)e * 4 + h] / (denom[(size_t)i * 4 + h] + 1e-16f);
    int d0 = lane * 8;
    const float* pl = xl + (size_t)j * HD + d0;
    float*       pm = m  + (size_t)i * HD + d0;
#pragma unroll
    for (int t = 0; t < 8; ++t) atomicAdd(&pm[t], pl[t] * sc);
}

// ---------------------------------------------------------------------------
// Bias-add + LayerNorm (in place). One wave per 256-wide row.
// ---------------------------------------------------------------------------
__global__ __launch_bounds__(256)
void ln_kernel(float* __restrict__ m, const float* __restrict__ bias,
               const float* __restrict__ g, const float* __restrict__ b, int N)
{
    int lane = threadIdx.x & 31;
    int row = blockIdx.x * 8 + (threadIdx.x >> 5);
    if (row >= N) return;
    float* pm = m + (size_t)row * HD;
    int d0 = lane * 8;
    float v[8];
    float s = 0.f, s2 = 0.f;
#pragma unroll
    for (int t = 0; t < 8; ++t) {
        float x = pm[d0 + t] + bias[d0 + t];
        v[t] = x; s += x; s2 += x * x;
    }
#pragma unroll
    for (int o = 1; o < 32; o <<= 1) {
        s  += __shfl_xor(s,  o);
        s2 += __shfl_xor(s2, o);
    }
    float mu  = s  * (1.0f / HD);
    float var = s2 * (1.0f / HD) - mu * mu;
    float inv = rsqrtf(var + 1e-5f);
#pragma unroll
    for (int t = 0; t < 8; ++t)
        pm[d0 + t] = (v[t] - mu) * inv * g[d0 + t] + b[d0 + t];
}

// ---------------------------------------------------------------------------
// GRU gates: h = (1-z)*tanh(i_n + r*h_n) + z*h_prev. One thread per (node,d).
// ---------------------------------------------------------------------------
__global__ void gru_gate_kernel(const float* __restrict__ gi,
                                const float* __restrict__ gh,
                                const float* __restrict__ hprev, int ldh,
                                float* __restrict__ out, int N)
{
    int t = blockIdx.x * blockDim.x + threadIdx.x;
    if (t >= N * OUTW) return;
    int nidx = t >> 6, d = t & 63;
    const float* pi = gi + (size_t)nidx * GW;
    const float* ph = gh + (size_t)nidx * GW;
    float hp = hprev[(size_t)nidx * ldh + d];
    float r  = 1.f / (1.f + expf(-(pi[d]      + ph[d])));
    float z  = 1.f / (1.f + expf(-(pi[64 + d] + ph[64 + d])));
    float nn = tanhf(pi[128 + d] + r * ph[128 + d]);
    out[(size_t)nidx * OUTW + d] = (1.f - z) * nn + z * hp;
}

// ---------------------------------------------------------------------------
// Host-side orchestration
// ---------------------------------------------------------------------------
template<int KIN, int NT>
static inline void launch_gemm_t(const float* X, int ldx, const _Float16* Wp,
                                 const float* bias, float* Y, int N,
                                 hipStream_t s)
{
    gemm_wmma_t<KIN, NT><<<N / 16, 128, 0, s>>>(X, ldx, Wp, bias, Y, NT * 64);
}

static inline void launch_pack(const float* W, int ldw, int transB,
                               _Float16* Wp, int Kin, int Kout, hipStream_t s)
{
    int total = (Kin >> 5) * (Kout >> 4) * 32;
    pack_w_kernel<<<(total + 255) / 256, 256, 0, s>>>(W, ldw, transB, Wp, Kin, Kout);
}

static inline void launch_fill(float* p, float v, long long n, hipStream_t s)
{
    long long blocks = (n + 255) / 256;
    fill_kernel<<<(unsigned)blocks, 256, 0, s>>>(p, v, n);
}

extern "C" void kernel_launch(void* const* d_in, const int* in_sizes, int n_in,
                              void* d_out, int out_size, void* d_ws, size_t ws_size,
                              hipStream_t stream)
{
    const float* x       = (const float*)d_in[0];
    const int*   ei      = (const int*)  d_in[1];   // (2, 2M) row-major
    const float* v2c_Wl  = (const float*)d_in[3];
    const float* v2c_Wr  = (const float*)d_in[4];
    const float* v2c_att = (const float*)d_in[5];
    const float* v2c_b   = (const float*)d_in[6];
    const float* c_ln_g  = (const float*)d_in[7];
    const float* c_ln_b  = (const float*)d_in[8];
    const float* c_Wih   = (const float*)d_in[9];   // (192,256)
    const float* c_Whh   = (const float*)d_in[10];  // (192,64)
    const float* c_bih   = (const float*)d_in[11];
    const float* c_bhh   = (const float*)d_in[12];
    const float* c2v_Wl  = (const float*)d_in[13];  // (64,256)
    const float* c2v_Wr  = (const float*)d_in[14];  // (96,256)
    const float* c2v_att = (const float*)d_in[15];
    const float* c2v_b   = (const float*)d_in[16];
    const float* v_ln_g  = (const float*)d_in[17];
    const float* v_ln_b  = (const float*)d_in[18];
    const float* v_Wih   = (const float*)d_in[19];
    const float* v_Whh   = (const float*)d_in[20];
    const float* v_bih   = (const float*)d_in[21];
    const float* v_bhh   = (const float*)d_in[22];
    float* out = (float*)d_out;                     // [V+C, 64]

    // Workspace carve-up
    float* ws   = (float*)d_ws;
    float* XL   = ws;                               // 400000*256
    float* XR   = XL   + (size_t)CN * HD;           // 400000*256
    float* M    = XR   + (size_t)CN * HD;           // 400000*256
    float* GI   = M    + (size_t)CN * HD;           // 400000*192
    float* GH   = GI   + (size_t)CN * GW;           // 400000*192
    float* AL   = GH   + (size_t)CN * GW;           // 1M * 4
    float* AMAX = AL   + (size_t)EHALF * 4;         // 400000*4
    float* DEN  = AMAX + (size_t)CN * 4;            // 400000*4

    _Float16* WP      = (_Float16*)(DEN + (size_t)CN * 4);
    _Float16* P_v2cWl = WP;                          // 96*256
    _Float16* P_v2cWr = P_v2cWl + 96 * 256;          // 96*256
    _Float16* P_cWih  = P_v2cWr + 96 * 256;          // 256*192
    _Float16* P_cWhh  = P_cWih  + 256 * 192;         // 64*192
    _Float16* P_c2vWl = P_cWhh  + 64 * 192;          // 64*256
    _Float16* P_c2vWr = P_c2vWl + 64 * 256;          // 96*256
    _Float16* P_vWih  = P_c2vWr + 96 * 256;          // 256*192
    _Float16* P_vWhh  = P_vWih  + 256 * 192;         // 64*192

    const int* src1 = ei;                  // v2c: j = src-V,  i = dst
    const int* dst1 = ei + ETOT;
    const int* src2 = ei + EHALF;          // c2v: j = src,    i = dst-V
    const int* dst2 = ei + ETOT + EHALF;

    const float* h_c_old = x + (size_t)VN * INW + 32;   // x[V:, 32:96], ld 96
    const float* h_v_old = x + 32;                      // x[:V, 32:96], ld 96
    float* h_c_new = out + (size_t)VN * OUTW;           // d_out[V:]
    float* h_v_new = out;                               // d_out[:V]

    // -------- weight pre-pack (f32 -> f16 WMMA fragments), once per call ----
    launch_pack(v2c_Wl, HD,   0, P_v2cWl, INW,  HD, stream);
    launch_pack(v2c_Wr, HD,   0, P_v2cWr, INW,  HD, stream);
    launch_pack(c_Wih,  HD,   1, P_cWih,  HD,   GW, stream);
    launch_pack(c_Whh,  OUTW, 1, P_cWhh,  OUTW, GW, stream);
    launch_pack(c2v_Wl, HD,   0, P_c2vWl, OUTW, HD, stream);
    launch_pack(c2v_Wr, HD,   0, P_c2vWr, INW,  HD, stream);
    launch_pack(v_Wih,  HD,   1, P_vWih,  HD,   GW, stream);
    launch_pack(v_Whh,  OUTW, 1, P_vWhh,  OUTW, GW, stream);

    // ======================= Stage 1: var -> clause =======================
    launch_fill(M,    0.f,    (long long)CN * HD, stream);
    launch_fill(AMAX, -1e30f, (long long)CN * 4,  stream);
    launch_fill(DEN,  0.f,    (long long)CN * 4,  stream);

    launch_gemm_t<INW, 4>(x, INW, P_v2cWl, nullptr, XL, VN, stream);
    launch_gemm_t<INW, 4>(x + (size_t)VN * INW, INW, P_v2cWr, nullptr, XR, CN, stream);

    edge_score_kernel<<<EHALF / 8, 256, 0, stream>>>(src1, VN, dst1, 0,
                                                     XL, XR, v2c_att, AL, AMAX, EHALF);
    edge_softmax_kernel<<<(EHALF * 4) / 256, 256, 0, stream>>>(dst1, 0, AL, AMAX, DEN, EHALF);
    edge_aggregate_kernel<<<EHALF / 8, 256, 0, stream>>>(src1, VN, dst1, 0,
                                                         XL, AL, DEN, M, EHALF);

    ln_kernel<<<CN / 8, 256, 0, stream>>>(M, v2c_b, c_ln_g, c_ln_b, CN);

    launch_gemm_t<HD,   3>(M, HD, P_cWih, c_bih, GI, CN, stream);
    launch_gemm_t<OUTW, 3>(h_c_old, INW, P_cWhh, c_bhh, GH, CN, stream);
    gru_gate_kernel<<<(CN * OUTW) / 256, 256, 0, stream>>>(GI, GH, h_c_old, INW,
                                                           h_c_new, CN);

    // ======================= Stage 2: clause -> var =======================
    launch_fill(M,    0.f,    (long long)VN * HD, stream);
    launch_fill(AMAX, -1e30f, (long long)VN * 4,  stream);
    launch_fill(DEN,  0.f,    (long long)VN * 4,  stream);

    launch_gemm_t<OUTW, 4>(h_c_new, OUTW, P_c2vWl, nullptr, XL, CN, stream);
    launch_gemm_t<INW,  4>(x, INW, P_c2vWr, nullptr, XR, VN, stream);

    edge_score_kernel<<<EHALF / 8, 256, 0, stream>>>(src2, 0, dst2, VN,
                                                     XL, XR, c2v_att, AL, AMAX, EHALF);
    edge_softmax_kernel<<<(EHALF * 4) / 256, 256, 0, stream>>>(dst2, VN, AL, AMAX, DEN, EHALF);
    edge_aggregate_kernel<<<EHALF / 8, 256, 0, stream>>>(src2, 0, dst2, VN,
                                                         XL, AL, DEN, M, EHALF);

    ln_kernel<<<VN / 8, 256, 0, stream>>>(M, c2v_b, v_ln_g, v_ln_b, VN);

    launch_gemm_t<HD,   3>(M, HD, P_vWih, v_bih, GI, VN, stream);
    launch_gemm_t<OUTW, 3>(h_v_old, INW, P_vWhh, v_bhh, GH, VN, stream);
    gru_gate_kernel<<<(VN * OUTW) / 256, 256, 0, stream>>>(GI, GH, h_v_old, INW,
                                                           h_v_new, VN);
}
